// HashLinear_73882027426043
// MI455X (gfx1250) — compile-verified, matrix operand order checked by reading, not run on hardware
//
#include <hip/hip_runtime.h>

// ---------------------------------------------------------------------------
// HashLinear on MI455X (gfx1250):
//   Kernel 1: materialize hashed weight pool -> f16 W[N][K] in workspace.
//   Kernel 2: f16 WMMA GEMM  out[M][N] = (A[M][K] @ W[N][K]^T + bias) * factor
//   M=8192, N=4096, K=4096.  AI ~ 2000 flop/byte -> compute-bound -> WMMA.
//   B (weight) tile staged via global_load_async_to_lds (ASYNCcnt) when the
//   toolchain exposes the gfx1250 async builtins; double-buffered LDS.
// ---------------------------------------------------------------------------

typedef _Float16 v16h __attribute__((ext_vector_type(16)));
typedef _Float16 v8h  __attribute__((ext_vector_type(8)));
typedef _Float16 v4h  __attribute__((ext_vector_type(4)));
typedef float    v8f  __attribute__((ext_vector_type(8)));
typedef float    vf4  __attribute__((ext_vector_type(4)));
typedef int      vi4  __attribute__((ext_vector_type(4)));

#define K_IN   4096
#define N_OUT  4096
#define POOL_HALF 524288u      // 1048576 / 2 floats
#define POOL_BLOCKS 8192u      // POOL_HALF / 64  (power of two -> AND)

#define BM 128
#define BN 256
#define BK 32
#define LDT 40                 // padded LDS row stride (halves): 20 words -> bank-conflict-free

#if defined(__gfx1250__) && \
    __has_builtin(__builtin_amdgcn_global_load_async_to_lds_b128) && \
    __has_builtin(__builtin_amdgcn_s_wait_asynccnt)
#define HASHW_ASYNC 1
// builtin signature: (int4 AS1* src_global, int4 AS3* dst_lds, imm offset, imm cpol)
typedef __attribute__((address_space(1))) vi4 gvi4;
typedef __attribute__((address_space(3))) vi4 svi4;
#else
#define HASHW_ASYNC 0
#endif

// ---------------------------------------------------------------------------
// Kernel 1: W_f16[n*K + k] = f16( 0.5*(pool1[h1(bid)*64+off] + pool2[h2(bid)*64+off]) )
// bid = flat/64; hash = bid*2654435761 ^ (seed*0x9E3779B9); idx = hash & 8191.
// Each thread produces 4 consecutive elements (one float4 from each pool half).
// ---------------------------------------------------------------------------
__global__ __launch_bounds__(256) void hashw_build_kernel(
    const float* __restrict__ pool, _Float16* __restrict__ Wh)
{
    const unsigned gid  = blockIdx.x * 256u + threadIdx.x;    // one per 4 elements
    const size_t   base = (size_t)gid * 4;                    // flat index into N*K
    const unsigned bid  = (unsigned)(base >> 6);
    const unsigned off  = (unsigned)base & 63u;

    const unsigned mixc = bid * 2654435761u;
    const unsigned h1   = (mixc ^ (1234u * 0x9E3779B9u)) & (POOL_BLOCKS - 1u);
    const unsigned h2   = (mixc ^ (1235u * 0x9E3779B9u)) & (POOL_BLOCKS - 1u);

    vf4 a = *(const vf4*)(pool + ((size_t)h1 << 6) + off);
    vf4 b = *(const vf4*)(pool + POOL_HALF + ((size_t)h2 << 6) + off);
    vf4 s = (a + b) * 0.5f;
    *(v4h*)(Wh + base) = __builtin_convertvector(s, v4h);
}

// ---------------------------------------------------------------------------
// Kernel 2: WMMA GEMM. 256 threads = 8 waves (2 along M x 4 along N).
// Block tile 128M x 256N, K step 32; wave tile 64M x 64N = 4x4 subtiles of
// 16x16 -> 16 v_wmma_f32_16x16x32_f16 per K-step per wave.
// A and W are both k-major per row, so A- and B-fragments use the identical
// per-lane pattern (CDNA5 16-bit 16x32 layout): lanes 0-15 hold row m=lane,
// K chunks {0..7,16..23}; lanes 16-31 hold the same row, chunks {8..15,24..31}.
// Each fragment = two 16-byte LDS loads. LDS is double buffered: next tile's
// fills (async for B) are issued before the current tile's WMMAs.
// ---------------------------------------------------------------------------
__global__ __launch_bounds__(256) void hash_gemm_kernel(
    const float*    __restrict__ A,      // [M][K] f32
    const _Float16* __restrict__ W,      // [N][K] f16
    const float*    __restrict__ bias,   // [N]
    const float*    __restrict__ factor, // [1]
    float*          __restrict__ out)    // [M][N]
{
    __shared__ __align__(16) _Float16 As[2][BM * LDT];   // 2 x 10 KB
    __shared__ __align__(16) _Float16 Bs[2][BN * LDT];   // 2 x 20 KB

    const int tid  = threadIdx.x;
    const int lane = tid & 31;
    const int wid  = tid >> 5;
    const int wm   = (wid & 1) * 64;     // wave M offset in tile
    const int wn   = (wid >> 1) * 64;    // wave N offset in tile

    const long blockN = (long)blockIdx.x * BN;
    const long blockM = (long)blockIdx.y * BM;

    // A-tile loader: 128 rows x 32 f32 -> f16. thread: 4 rows, one float4 each.
    const int a_kq = (tid & 7) * 4;      // f32 quad within row
    const int a_r0 = tid >> 3;           // row 0..31 (+32 stride x4)
    // B-tile loader: 256 rows x 32 f16. thread: 4 rows, one 16B chunk each.
    const int b_kh = (tid & 3) * 8;      // f16 octet within row
    const int b_r0 = tid >> 2;           // row 0..63 (+64 stride x4)

    // fragment addressing (CDNA5 16-bit 16x32 layout)
    const int frow = lane & 15;
    const int c0   = (lane < 16) ? 0 : 8;

    v8f acc[4][4] = {};

    auto load_tiles = [&](int b, int kt) {
        #pragma unroll
        for (int rr = 0; rr < 4; ++rr) {
            const int row = a_r0 + rr * 32;
            vf4 x = *(const vf4*)(A + (blockM + row) * (long)K_IN + kt + a_kq);
            *(v4h*)(&As[b][row * LDT + a_kq]) = __builtin_convertvector(x, v4h);
        }
        #pragma unroll
        for (int rr = 0; rr < 4; ++rr) {
            const int row = b_r0 + rr * 64;
            const _Float16* src = W + (blockN + row) * (long)K_IN + kt + b_kh;
            _Float16*       dst = &Bs[b][row * LDT + b_kh];
#if HASHW_ASYNC
            // gfx1250: DMA 16B straight into LDS, tracked by ASYNCcnt.
            __builtin_amdgcn_global_load_async_to_lds_b128(
                (gvi4*)src, (svi4*)dst, 0, 0);
#else
            *(v8h*)dst = *(const v8h*)src;
#endif
        }
    };

    int buf = 0;
    load_tiles(0, 0);

    for (int kt = 0; kt < K_IN; kt += BK) {
#if HASHW_ASYNC
        __builtin_amdgcn_s_wait_asynccnt(0);   // drain this wave's async fills
#endif
        __syncthreads();                       // all waves' tile[buf] visible

        if (kt + BK < K_IN)
            load_tiles(buf ^ 1, kt + BK);      // prefetch next tile (other buffer)

        union H16 { v16h v; v8h h[2]; };
        v16h af[4], bf[4];
        #pragma unroll
        for (int i = 0; i < 4; ++i) {
            const int r = wm + i * 16 + frow;
            H16 u;
            u.h[0] = *(const v8h*)(&As[buf][r * LDT + c0]);
            u.h[1] = *(const v8h*)(&As[buf][r * LDT + c0 + 16]);
            af[i] = u.v;
        }
        #pragma unroll
        for (int j = 0; j < 4; ++j) {
            const int r = wn + j * 16 + frow;
            H16 u;
            u.h[0] = *(const v8h*)(&Bs[buf][r * LDT + c0]);
            u.h[1] = *(const v8h*)(&Bs[buf][r * LDT + c0 + 16]);
            bf[j] = u.v;
        }

        #pragma unroll
        for (int i = 0; i < 4; ++i)
            #pragma unroll
            for (int j = 0; j < 4; ++j)
                acc[i][j] = __builtin_amdgcn_wmma_f32_16x16x32_f16(
                    false, af[i], false, bf[j],
                    (short)0, acc[i][j], false, false);

        buf ^= 1;
    }

    // Epilogue: C/D layout -> VGPR r holds M = m0 + r (lanes 0-15) and
    // M = m0 + 8 + r (lanes 16-31); N = lane & 15.
    const float f = factor[0];
    #pragma unroll
    for (int j = 0; j < 4; ++j) {
        const long col = blockN + wn + j * 16 + (lane & 15);
        const float bv = bias[col] * f;
        #pragma unroll
        for (int i = 0; i < 4; ++i) {
            const long rbase = blockM + wm + i * 16 + ((lane >> 4) << 3);
            #pragma unroll
            for (int r = 0; r < 8; ++r)
                out[(rbase + r) * (long)N_OUT + col] = acc[i][j][r] * f + bv;
        }
    }
}

// ---------------------------------------------------------------------------
extern "C" void kernel_launch(void* const* d_in, const int* in_sizes, int n_in,
                              void* d_out, int out_size, void* d_ws, size_t ws_size,
                              hipStream_t stream)
{
    const float* input  = (const float*)d_in[0];   // [4,2048,4096] f32
    const float* pool   = (const float*)d_in[1];   // [1048576] f32
    const float* bias   = (const float*)d_in[2];   // [4096] f32
    const float* factor = (const float*)d_in[3];   // [1] f32
    float*       out    = (float*)d_out;           // [M][4096] f32
    _Float16*    Wh     = (_Float16*)d_ws;         // 4096*4096 f16 = 32 MB scratch

    const int M = in_sizes[0] / K_IN;              // 8192

    // 1) hashed-weight materialization: (N*K)/4 threads
    const int nthreads1 = (N_OUT * K_IN) / 4;
    hashw_build_kernel<<<nthreads1 / 256, 256, 0, stream>>>(pool, Wh);

    // 2) WMMA GEMM
    dim3 grid(N_OUT / BN, M / BM);
    hash_gemm_kernel<<<grid, 256, 0, stream>>>(input, Wh, bias, factor, out);
}
